// NeRF_59004260712587
// MI455X (gfx1250) — matrix operand order repaired and checked by reading, not compiled, not run on hardware
//
#include <hip/hip_runtime.h>
#include <stdint.h>

typedef __attribute__((ext_vector_type(16))) _Float16 v16h;
typedef __attribute__((ext_vector_type(8)))  _Float16 v8h;
typedef __attribute__((ext_vector_type(8)))  float    v8f;

#define WMMA(a,b,c) __builtin_amdgcn_wmma_f32_16x16x32_f16(false,(a),false,(b),(short)0,(c),false,false)

namespace {
constexpr int TSIZE = 1 << 19;
constexpr int KPS = 72;    // padded row stride (halves) for K<=64 weight/act rows
constexpr int KPL = 136;   // padded row stride (halves) for K<=128 rows
constexpr int W_S0 = 0;                      // 64 x 72
constexpr int W_S1 = W_S0 + 64 * KPS;        // 16 x 72
constexpr int W_C0 = W_S1 + 16 * KPS;        // 64 x 136
constexpr int W_C1 = W_C0 + 64 * KPL;        // 64 x 72
constexpr int W_C2 = W_C1 + 64 * KPS;        // 16 x 72
constexpr int ACT0 = W_C2 + 16 * KPS;        // 8 waves x 16 x 136
constexpr int ACTW = 16 * KPL;
constexpr int HID0 = ACT0 + 8 * ACTW;        // 8 waves x 16 x 72
constexpr int HIDW = 16 * KPS;
constexpr int SMEM_H = HID0 + 8 * HIDW;      // 46848 halves = 93696 B

__constant__ float RES_LO[8] = {16.f, 22.f, 30.f, 42.f, 58.f, 80.f, 111.f, 153.f};
__constant__ float RES_HI[8] = {212.f, 294.f, 406.f, 561.f, 776.f, 1072.f, 1482.f, 2048.f};
}

__device__ __forceinline__ v16h frag16(const _Float16* p0, const _Float16* p1) {
  v8h lo = *(const v8h*)p0;
  v8h hh = *(const v8h*)p1;
  return __builtin_shufflevector(lo, hh, 0,1,2,3,4,5,6,7,8,9,10,11,12,13,14,15);
}
// A fragment: activations [row=point][k], CDNA5 16-bit A layout
__device__ __forceinline__ v16h load_a(const _Float16* s, int base, int stride, int row, int kc, int hi) {
  const int off = base + row * stride + kc * 32 + (hi ? 8 : 0);
  return frag16(s + off, s + off + 16);
}
// B fragment: weights [out][k]; lane holds one output column, same K order as A
__device__ __forceinline__ v16h load_b(const _Float16* s, int base, int stride, int col, int kc, int hi) {
  const int off = base + col * stride + kc * 32 + (hi ? 8 : 0);
  return frag16(s + off, s + off + 16);
}

__global__ __launch_bounds__(256) void nerf_fused(
    const float* __restrict__ xin, const float* __restrict__ ht,
    const float* __restrict__ planes, const float* __restrict__ embed,
    const float* __restrict__ ws0, const float* __restrict__ ws1,
    const float* __restrict__ wc0, const float* __restrict__ wc1,
    const float* __restrict__ wc2, float* __restrict__ out, int npts)
{
  __shared__ _Float16 smem[SMEM_H];
  const int tid = threadIdx.x;

  { // zero all LDS (pads must be zero)
    uint32_t* z = (uint32_t*)smem;
    for (int i = tid; i < SMEM_H / 2; i += 256) z[i] = 0u;
  }
  __syncthreads();
  // stage f16 weights; w_color0 columns permuted: [sh16, geo15, pad1, app48, pf24]
  for (int i = tid; i < 64 * 56; i += 256) { int r = i / 56, k = i - r * 56; smem[W_S0 + r * KPS + k] = (_Float16)ws0[i]; }
  for (int i = tid; i < 16 * 64; i += 256) { int r = i >> 6, k = i & 63; smem[W_S1 + r * KPS + k] = (_Float16)ws1[i]; }
  for (int i = tid; i < 64 * 103; i += 256) {
    int r = i / 103, k = i - r * 103; int kp = (k < 31) ? k : k + 1;
    smem[W_C0 + r * KPL + kp] = (_Float16)wc0[i];
  }
  for (int i = tid; i < 64 * 64; i += 256) { int r = i >> 6, k = i & 63; smem[W_C1 + r * KPS + k] = (_Float16)wc1[i]; }
  for (int i = tid; i < 3 * 64;  i += 256) { int r = i >> 6, k = i & 63; smem[W_C2 + r * KPS + k] = (_Float16)wc2[i]; }
  __syncthreads();

  const int wv = tid >> 5, lane = tid & 31, row = lane & 15;
  const int hi = lane >> 4, mb = hi ? 8 : 0;
  const int ACTB = ACT0 + wv * ACTW, HIDB = HID0 + wv * HIDW;

  #pragma unroll 1
  for (int t = 0; t < 2; ++t) {
    const int pbase = blockIdx.x * 256 + (wv * 2 + t) * 16;
    if (pbase >= npts) continue;
    const int p = pbase + row;
    const float* xr = xin + (size_t)p * 7;
    const float p0x = (xr[0] + 1.f) * 0.5f;
    const float p0y = (xr[1] + 1.f) * 0.5f;
    const float p0z = (xr[2] + 1.f) * 0.5f;
    const float dxr = xr[3], dyr = xr[4], dzr = xr[5];
    const int aidx = (int)xr[6];

    // ---- hash grid: lo half-wave does levels 0..7, hi half does 8..15 ----
    {
      const int hb = ACTB + row * KPL + (hi ? 16 : 0);
      #pragma unroll
      for (int lv = 0; lv < 8; ++lv) {
        const float R = hi ? RES_HI[lv] : RES_LO[lv];
        const int lvl = lv + mb;
        const float px = p0x * R, py = p0y * R, pz = p0z * R;
        const float fx = floorf(px), fy = floorf(py), fz = floorf(pz);
        const uint32_t ix = (uint32_t)fx, iy = (uint32_t)fy, iz = (uint32_t)fz;
        const float rx = px - fx, ry = py - fy, rz = pz - fz;
        const float* tb = ht + ((size_t)lvl << 20);   // T*F = 2^20 floats/level
        float f0 = 0.f, f1 = 0.f;
        #pragma unroll
        for (int c = 0; c < 8; ++c) {
          const uint32_t b0 = (c >> 2) & 1, b1 = (c >> 1) & 1, b2 = c & 1;
          uint32_t idx = ((ix + b0) * 1u ^ (iy + b1) * 2654435761u ^ (iz + b2) * 805459861u)
                         & (uint32_t)(TSIZE - 1);
          const float w = (b0 ? rx : 1.f - rx) * (b1 ? ry : 1.f - ry) * (b2 ? rz : 1.f - rz);
          const float2 f = *(const float2*)(tb + ((size_t)idx << 1));
          f0 += w * f.x; f1 += w * f.y;
        }
        smem[hb + 2 * lv]     = (_Float16)f0;
        smem[hb + 2 * lv + 1] = (_Float16)f1;
      }
    }

    // ---- tri-planes: lo half does planes 0,1; hi half does plane 2 ----
    {
      const int nrep = hi ? 1 : 2;
      #pragma unroll 1
      for (int rep = 0; rep < nrep; ++rep) {
        const int pi = hi ? 2 : rep;
        const float uc = (pi == 2 ? p0y : p0x) * 511.f;
        const float vc = (pi == 0 ? p0y : p0z) * 511.f;
        const float uf = floorf(uc), vf = floorf(vc);
        const float wu = uc - uf, wv2 = vc - vf;
        int ui = (int)uf; ui = ui < 0 ? 0 : (ui > 510 ? 510 : ui);
        int vi = (int)vf; vi = vi < 0 ? 0 : (vi > 510 ? 510 : vi);
        const float* pl = planes + ((size_t)pi * 512 * 512 + (size_t)ui * 512 + vi) * 8;
        const float w00 = (1.f - wu) * (1.f - wv2), w01 = (1.f - wu) * wv2;
        const float w10 = wu * (1.f - wv2),         w11 = wu * wv2;
        v8h o;
        #pragma unroll
        for (int j = 0; j < 8; ++j) {
          float f = pl[j] * w00 + pl[8 + j] * w01 + pl[4096 + j] * w10 + pl[4096 + 8 + j] * w11;
          o[j] = (_Float16)f;
        }
        *(v8h*)(smem + ACTB + row * KPL + 32 + pi * 8) = o;
      }
    }
    // zero sigma-input pad cols 56..63
    if (!hi) { v8h z = {}; *(v8h*)(smem + ACTB + row * KPL + 56) = z; }

    // ---- sigma layer 0: [16x64] x [64x64] ----
    v8f a0 = {}, a1 = {}, a2 = {}, a3 = {};
    #pragma unroll
    for (int kc = 0; kc < 2; ++kc) {
      v16h a = load_a(smem, ACTB, KPL, row, kc, hi);
      a0 = WMMA(a, load_b(smem, W_S0, KPS,      row, kc, hi), a0);
      a1 = WMMA(a, load_b(smem, W_S0, KPS, 16 + row, kc, hi), a1);
      a2 = WMMA(a, load_b(smem, W_S0, KPS, 32 + row, kc, hi), a2);
      a3 = WMMA(a, load_b(smem, W_S0, KPS, 48 + row, kc, hi), a3);
    }
    #pragma unroll
    for (int v = 0; v < 8; ++v) {
      const int rb = HIDB + (v + mb) * KPS;
      float r0 = a0[v], r1 = a1[v], r2 = a2[v], r3 = a3[v];
      smem[rb + row]      = (_Float16)(r0 > 0.f ? r0 : 0.f);
      smem[rb + 16 + row] = (_Float16)(r1 > 0.f ? r1 : 0.f);
      smem[rb + 32 + row] = (_Float16)(r2 > 0.f ? r2 : 0.f);
      smem[rb + 48 + row] = (_Float16)(r3 > 0.f ? r3 : 0.f);
    }

    // ---- sigma layer 1: [16x64] x [64x16] ----
    v8f sg = {};
    #pragma unroll
    for (int kc = 0; kc < 2; ++kc) {
      v16h a = load_a(smem, HIDB, KPS, row, kc, hi);
      sg = WMMA(a, load_b(smem, W_S1, KPS, row, kc, hi), sg);
    }
    if (row == 0) {
      #pragma unroll
      for (int v = 0; v < 8; ++v) {
        float h0 = fminf(fmaxf(sg[v], -15.f), 15.f);
        out[(size_t)(pbase + v + mb) * 4 + 3] = __expf(h0);
        smem[ACTB + (v + mb) * KPL + 31] = (_Float16)0.f;   // pad column
      }
    } else {
      #pragma unroll
      for (int v = 0; v < 8; ++v)
        smem[ACTB + (v + mb) * KPL + 15 + row] = (_Float16)sg[v];  // geo -> cols 16..30
    }

    // ---- build color input: [sh16 | geo15 | pad | app48 | pf24 | zeros] ----
    _Float16 pfv[12];
    {
      const int so = ACTB + row * KPL + 32 + (hi ? 12 : 0);  // pf lives in cols 32..55
      #pragma unroll
      for (int j = 0; j < 12; ++j) pfv[j] = smem[so + j];
    }
    {
      const float inv = rsqrtf(dxr * dxr + dyr * dyr + dzr * dzr);
      const float x = dxr * inv, y = dyr * inv, z = dzr * inv;
      const float xx = x * x, yy = y * y, zz = z * z, xy = x * y, yz = y * z, xz = x * z;
      float s[8];
      if (!hi) {
        s[0] = 0.28209479177387814f;           s[1] = -0.48860251190291987f * y;
        s[2] = 0.48860251190291987f * z;       s[3] = -0.48860251190291987f * x;
        s[4] = 1.0925484305920792f * xy;       s[5] = -1.0925484305920792f * yz;
        s[6] = 0.94617469575756f * zz - 0.31539156525252005f;
        s[7] = -1.0925484305920792f * xz;
      } else {
        s[0] = 0.5462742152960396f * (xx - yy);
        s[1] = -0.5900435899266435f * y * (3.f * xx - yy);
        s[2] = 2.890611442640554f * xy * z;
        s[3] = -0.4570457994644658f * y * (4.f * zz - xx - yy);
        s[4] = 0.3731763325901154f * z * (2.f * zz - 3.f * xx - 3.f * yy);
        s[5] = -0.4570457994644658f * x * (4.f * zz - xx - yy);
        s[6] = 1.445305721320277f * z * (xx - yy);
        s[7] = -0.5900435899266435f * x * (xx - 3.f * yy);
      }
      const int sb = ACTB + row * KPL + (hi ? 8 : 0);
      #pragma unroll
      for (int j = 0; j < 8; ++j) smem[sb + j] = (_Float16)s[j];
    }
    {
      const float* av = embed + (size_t)aidx * 48 + (hi ? 24 : 0);
      const int ab = ACTB + row * KPL + 32 + (hi ? 24 : 0);   // app -> cols 32..79
      #pragma unroll
      for (int j = 0; j < 24; ++j) smem[ab + j] = (_Float16)av[j];
    }
    {
      const int pb2 = ACTB + row * KPL + 80 + (hi ? 12 : 0);  // pf -> cols 80..103
      #pragma unroll
      for (int j = 0; j < 12; ++j) smem[pb2 + j] = pfv[j];
    }

    // ---- color layer 0: [16x128] x [128x64] ----
    v8f c0 = {}, c1 = {}, c2 = {}, c3 = {};
    #pragma unroll
    for (int kc = 0; kc < 4; ++kc) {
      v16h a = load_a(smem, ACTB, KPL, row, kc, hi);
      c0 = WMMA(a, load_b(smem, W_C0, KPL,      row, kc, hi), c0);
      c1 = WMMA(a, load_b(smem, W_C0, KPL, 16 + row, kc, hi), c1);
      c2 = WMMA(a, load_b(smem, W_C0, KPL, 32 + row, kc, hi), c2);
      c3 = WMMA(a, load_b(smem, W_C0, KPL, 48 + row, kc, hi), c3);
    }
    #pragma unroll
    for (int v = 0; v < 8; ++v) {
      const int rb = HIDB + (v + mb) * KPS;
      float r0 = c0[v], r1 = c1[v], r2 = c2[v], r3 = c3[v];
      smem[rb + row]      = (_Float16)(r0 > 0.f ? r0 : 0.f);
      smem[rb + 16 + row] = (_Float16)(r1 > 0.f ? r1 : 0.f);
      smem[rb + 32 + row] = (_Float16)(r2 > 0.f ? r2 : 0.f);
      smem[rb + 48 + row] = (_Float16)(r3 > 0.f ? r3 : 0.f);
    }

    // ---- color layer 1: [16x64] x [64x64] (A read before HID overwritten) ----
    {
      v16h b0 = load_a(smem, HIDB, KPS, row, 0, hi);
      v16h b1 = load_a(smem, HIDB, KPS, row, 1, hi);
      v8f d0 = {}, d1 = {}, d2 = {}, d3 = {};
      d0 = WMMA(b0, load_b(smem, W_C1, KPS,      row, 0, hi), d0);
      d1 = WMMA(b0, load_b(smem, W_C1, KPS, 16 + row, 0, hi), d1);
      d2 = WMMA(b0, load_b(smem, W_C1, KPS, 32 + row, 0, hi), d2);
      d3 = WMMA(b0, load_b(smem, W_C1, KPS, 48 + row, 0, hi), d3);
      d0 = WMMA(b1, load_b(smem, W_C1, KPS,      row, 1, hi), d0);
      d1 = WMMA(b1, load_b(smem, W_C1, KPS, 16 + row, 1, hi), d1);
      d2 = WMMA(b1, load_b(smem, W_C1, KPS, 32 + row, 1, hi), d2);
      d3 = WMMA(b1, load_b(smem, W_C1, KPS, 48 + row, 1, hi), d3);
      #pragma unroll
      for (int v = 0; v < 8; ++v) {
        const int rb = HIDB + (v + mb) * KPS;
        float r0 = d0[v], r1 = d1[v], r2 = d2[v], r3 = d3[v];
        smem[rb + row]      = (_Float16)(r0 > 0.f ? r0 : 0.f);
        smem[rb + 16 + row] = (_Float16)(r1 > 0.f ? r1 : 0.f);
        smem[rb + 32 + row] = (_Float16)(r2 > 0.f ? r2 : 0.f);
        smem[rb + 48 + row] = (_Float16)(r3 > 0.f ? r3 : 0.f);
      }
    }

    // ---- color layer 2: [16x64] x [64x16(3 real)] + sigmoid ----
    {
      v16h e0 = load_a(smem, HIDB, KPS, row, 0, hi);
      v16h e1 = load_a(smem, HIDB, KPS, row, 1, hi);
      v8f cc = {};
      cc = WMMA(e0, load_b(smem, W_C2, KPS, row, 0, hi), cc);
      cc = WMMA(e1, load_b(smem, W_C2, KPS, row, 1, hi), cc);
      if (row < 3) {
        #pragma unroll
        for (int v = 0; v < 8; ++v) {
          float val = cc[v];
          out[(size_t)(pbase + v + mb) * 4 + row] = 1.f / (1.f + __expf(-val));
        }
      }
    }
  }
}

extern "C" void kernel_launch(void* const* d_in, const int* in_sizes, int n_in,
                              void* d_out, int out_size, void* d_ws, size_t ws_size,
                              hipStream_t stream) {
  (void)n_in; (void)out_size; (void)d_ws; (void)ws_size;
  const float* x   = (const float*)d_in[0];
  const float* ht  = (const float*)d_in[1];
  const float* pl  = (const float*)d_in[2];
  const float* em  = (const float*)d_in[3];
  const float* ws0 = (const float*)d_in[4];
  const float* ws1 = (const float*)d_in[5];
  const float* wc0 = (const float*)d_in[6];
  const float* wc1 = (const float*)d_in[7];
  const float* wc2 = (const float*)d_in[8];
  float* out = (float*)d_out;
  const int n = in_sizes[0] / 7;
  const int grid = (n + 255) / 256;
  hipLaunchKernelGGL(nerf_fused, dim3(grid), dim3(256), 0, stream,
                     x, ht, pl, em, ws0, ws1, wc0, wc1, wc2, out, n);
}
